// DeepSpeedMLP_49143015801365
// MI455X (gfx1250) — matrix operand order repaired and checked by reading, not compiled, not run on hardware
//
#include <hip/hip_runtime.h>
#include <math.h>
#include <stdint.h>

#define M_TOK 4096
#define H_DIM 4096
#define DFF_DIM 16384

typedef __attribute__((ext_vector_type(16))) __bf16          v16bf;
typedef __attribute__((ext_vector_type(8)))  float           v8f;
typedef __attribute__((ext_vector_type(8)))  unsigned short  ush8;
typedef __attribute__((ext_vector_type(4)))  unsigned short  ush4;

// fp32 -> bf16, round-to-nearest-even
__device__ __forceinline__ unsigned short f2b(float f) {
  union { float f; unsigned u; } c; c.f = f;
  unsigned u = c.u;
  u += 0x7FFFu + ((u >> 16) & 1u);
  return (unsigned short)(u >> 16);
}

// tanh-GeLU via logistic identity: 0.5x(1+tanh(z)) = x * sigmoid(2z)
__device__ __forceinline__ float gelu_fast(float x) {
  float z2 = 1.5957691216057308f * (x + 0.044715f * x * x * x);
  return x * (1.0f / (1.0f + __expf(-z2)));
}

// ---------------------------------------------------------------------------
// Kernel 1: ln_in = input + residual + bias; LayerNorm(ln_in)*gamma+beta -> bf16
// ---------------------------------------------------------------------------
__global__ __launch_bounds__(256) void ln_bias_res_kernel(
    const float* __restrict__ input, const float* __restrict__ residual,
    const float* __restrict__ bias, const float* __restrict__ gamma,
    const float* __restrict__ beta, unsigned short* __restrict__ lnOut) {
  const int tid  = threadIdx.x;
  const int row  = blockIdx.x;
  const size_t rbase = (size_t)row * H_DIM;

  float4 v[4];
  float sum = 0.f, sq = 0.f;
#pragma unroll
  for (int i = 0; i < 4; ++i) {
    int c = (tid + i * 256) * 4;
    float4 a = *(const float4*)(input + rbase + c);
    float4 r = *(const float4*)(residual + rbase + c);
    float4 b = *(const float4*)(bias + c);
    float4 x;
    x.x = a.x + r.x + b.x;
    x.y = a.y + r.y + b.y;
    x.z = a.z + r.z + b.z;
    x.w = a.w + r.w + b.w;
    v[i] = x;
    sum += x.x + x.y + x.z + x.w;
    sq  += x.x * x.x + x.y * x.y + x.z * x.z + x.w * x.w;
  }
#pragma unroll
  for (int off = 16; off > 0; off >>= 1) {
    sum += __shfl_xor(sum, off, 32);
    sq  += __shfl_xor(sq,  off, 32);
  }
  __shared__ float s1[8], s2[8];
  __shared__ float stat[2];
  const int lane = tid & 31, wv = tid >> 5;
  if (lane == 0) { s1[wv] = sum; s2[wv] = sq; }
  __syncthreads();
  if (tid == 0) {
    float S = 0.f, Q = 0.f;
#pragma unroll
    for (int w = 0; w < 8; ++w) { S += s1[w]; Q += s2[w]; }
    float mu  = S * (1.0f / H_DIM);
    float var = Q * (1.0f / H_DIM) - mu * mu;
    stat[0] = mu;
    stat[1] = rsqrtf(var + 1e-5f);
  }
  __syncthreads();
  const float mu = stat[0], rstd = stat[1];
#pragma unroll
  for (int i = 0; i < 4; ++i) {
    int c = (tid + i * 256) * 4;
    float4 g  = *(const float4*)(gamma + c);
    float4 bb = *(const float4*)(beta + c);
    float4 x  = v[i];
    ush4 o;
    o[0] = f2b((x.x - mu) * rstd * g.x + bb.x);
    o[1] = f2b((x.y - mu) * rstd * g.y + bb.y);
    o[2] = f2b((x.z - mu) * rstd * g.z + bb.z);
    o[3] = f2b((x.w - mu) * rstd * g.w + bb.w);
    *(ush4*)(lnOut + rbase + c) = o;
  }
}

// ---------------------------------------------------------------------------
// Kernel 2: weight transpose + fp32->bf16:  in[R][C] f32  ->  out[C][R] bf16
// ---------------------------------------------------------------------------
__global__ __launch_bounds__(256) void transpose_convert_kernel(
    const float* __restrict__ in, unsigned short* __restrict__ out,
    int R, int C) {
  __shared__ unsigned short t[32][33];
  const int tx = threadIdx.x & 31, ty = threadIdx.x >> 5;
  const int r0 = blockIdx.y * 32, c0 = blockIdx.x * 32;
#pragma unroll
  for (int i = 0; i < 4; ++i) {
    int r = ty + i * 8;
    t[r][tx] = f2b(in[(size_t)(r0 + r) * C + (c0 + tx)]);
  }
  __syncthreads();
#pragma unroll
  for (int i = 0; i < 4; ++i) {
    int cc = ty + i * 8;
    out[(size_t)(c0 + cc) * R + (r0 + tx)] = t[tx][cc];
  }
}

// ---------------------------------------------------------------------------
// WMMA GEMM, 3-stage async pipeline (2 tiles in flight, counted asynccnt wait)
// C[M,N] = A[M,K](bf16) * Bt[N,K](bf16 pre-transposed)
// block tile 128x128, K-tile 64; 8 waves, each 64x32 (4x2 WMMA tiles)
// ---------------------------------------------------------------------------
#define KT     64
#define PITCH  72                    // halves; rows stay 16B aligned (144B)
#define TILE_H (128 * PITCH)         // halves per LDS buffer

template <bool GELU>
__global__ __launch_bounds__(256, 1) void wmma_gemm_async_kernel(
    const unsigned short* __restrict__ A,    // [M x K] bf16
    const unsigned short* __restrict__ Bt,   // [N x K] bf16
    const float* __restrict__ biasN, int K, int N,
    unsigned short* __restrict__ outB, float* __restrict__ outF,
    const float* __restrict__ add1, const float* __restrict__ add2,
    const float* __restrict__ biasX) {
  __shared__ unsigned short As[3 * TILE_H];
  __shared__ unsigned short Bs[3 * TILE_H];

  const int tid  = threadIdx.x;
  const int bm   = blockIdx.y * 128;
  const int bn   = blockIdx.x * 128;
  const int lane = tid & 31, wave = tid >> 5;
  const int wm   = wave & 1;
  const int wn   = wave >> 1;
  const int l16  = lane & 15;
  const int hf   = lane >> 4;

  // tile = 128 rows x 64 halves = 1024 x 16B chunks; thread moves chunks
  // {tid, tid+256} for A and for B via async DMA to LDS
  const int c0 = tid, c1 = tid + 256;
  const int r0 = c0 >> 3, q0 = (c0 & 7) * 8;
  const int r1 = c1 >> 3, q1 = (c1 & 7) * 8;
  const unsigned lds0 = (unsigned)((r0 * PITCH + q0) * 2);
  const unsigned lds1 = (unsigned)((r1 * PITCH + q1) * 2);
  const unsigned asB  = (unsigned)(uintptr_t)&As[0];   // low 32 bits = LDS addr
  const unsigned bsB  = (unsigned)(uintptr_t)&Bs[0];
  const unsigned BUFB = (unsigned)(TILE_H * 2);
  const unsigned gA0 = ((unsigned)(bm + r0) * (unsigned)K + (unsigned)q0) * 2u;
  const unsigned gA1 = ((unsigned)(bm + r1) * (unsigned)K + (unsigned)q1) * 2u;
  const unsigned gB0 = ((unsigned)(bn + r0) * (unsigned)K + (unsigned)q0) * 2u;
  const unsigned gB1 = ((unsigned)(bn + r1) * (unsigned)K + (unsigned)q1) * 2u;
  const uint64_t aP = (uint64_t)(uintptr_t)A;
  const uint64_t bP = (uint64_t)(uintptr_t)Bt;

#define ISSUE_TILE(bufByte, kByte)                                             \
  asm volatile(                                                                \
      "global_load_async_to_lds_b128 %0, %4, %8 offset:0\n\t"                  \
      "global_load_async_to_lds_b128 %1, %5, %8 offset:0\n\t"                  \
      "global_load_async_to_lds_b128 %2, %6, %9 offset:0\n\t"                  \
      "global_load_async_to_lds_b128 %3, %7, %9 offset:0"                      \
      :                                                                        \
      : "v"(asB + (bufByte) + lds0), "v"(asB + (bufByte) + lds1),              \
        "v"(bsB + (bufByte) + lds0), "v"(bsB + (bufByte) + lds1),              \
        "v"(gA0 + (kByte)), "v"(gA1 + (kByte)),                                \
        "v"(gB0 + (kByte)), "v"(gB1 + (kByte)),                                \
        "s"(aP), "s"(bP)                                                       \
      : "memory")

  v8f acc[4][2];
#pragma unroll
  for (int mi = 0; mi < 4; ++mi)
#pragma unroll
    for (int ni = 0; ni < 2; ++ni)
#pragma unroll
      for (int j = 0; j < 8; ++j) acc[mi][ni][j] = 0.0f;

  // one K-tile of WMMA work from LDS buffer at half-offset curH
  auto compute_tile = [&](unsigned curH) {
    const unsigned short* Ac = &As[curH];
    const unsigned short* Bc = &Bs[curH];
#pragma unroll
    for (int ks = 0; ks < 2; ++ks) {
      const int kb = ks * 32;
      union Frag { v16bf v; ush8 h[2]; };
      Frag af[4], bfr[2];
#pragma unroll
      for (int mi = 0; mi < 4; ++mi) {
        const unsigned short* p = &Ac[(wm * 64 + mi * 16 + l16) * PITCH + kb + hf * 8];
        af[mi].h[0] = *(const ush8*)p;
        af[mi].h[1] = *(const ush8*)(p + 16);
      }
#pragma unroll
      for (int ni = 0; ni < 2; ++ni) {
        const unsigned short* p = &Bc[(wn * 32 + ni * 16 + l16) * PITCH + kb + hf * 8];
        bfr[ni].h[0] = *(const ush8*)p;
        bfr[ni].h[1] = *(const ush8*)(p + 16);
      }
#pragma unroll
      for (int mi = 0; mi < 4; ++mi)
#pragma unroll
        for (int ni = 0; ni < 2; ++ni)
          acc[mi][ni] = __builtin_amdgcn_wmma_f32_16x16x32_bf16(
              false, af[mi].v, false, bfr[ni].v, (short)0, acc[mi][ni],
              false, false);
    }
  };

  const int nk = K / KT;   // 64 or 256 here (>= 3)
  // prologue: 2 tiles in flight
  ISSUE_TILE(0u, 0u);
  ISSUE_TILE(BUFB, (unsigned)(KT * 2));
  asm volatile("s_wait_asynccnt 0x4" ::: "memory");   // tile 0 done (in-order)
  __syncthreads();

  unsigned curH = 0;            // LDS half-offset of tile being computed
  unsigned preB = 2u * BUFB;    // LDS byte-offset of tile being prefetched
  unsigned kPre = (unsigned)(2 * KT * 2);
  // steady state: branch-free body, always one prefetch in flight
  for (int t = 0; t < nk - 2; ++t) {
    ISSUE_TILE(preB, kPre);
    compute_tile(curH);
    asm volatile("s_wait_asynccnt 0x4" ::: "memory");
    __syncthreads();
    curH += TILE_H; if (curH == 3u * TILE_H) curH = 0;
    preB += BUFB;   if (preB == 3u * BUFB)   preB = 0;
    kPre += (unsigned)(KT * 2);
  }
  // epilogue tiles: nk-2 (drain last DMA), then nk-1
  compute_tile(curH);
  asm volatile("s_wait_asynccnt 0x0" ::: "memory");
  __syncthreads();
  curH += TILE_H; if (curH == 3u * TILE_H) curH = 0;
  compute_tile(curH);
#undef ISSUE_TILE

  // ---- epilogue: C layout = lane l16 -> N, VGPR j + 8*hf -> M ----
#pragma unroll
  for (int mi = 0; mi < 4; ++mi) {
#pragma unroll
    for (int ni = 0; ni < 2; ++ni) {
      const int n  = bn + wn * 32 + ni * 16 + l16;
      const int m0 = bm + wm * 64 + mi * 16 + hf * 8;
      const float bv = biasN[n];
      if (GELU) {
#pragma unroll
        for (int j = 0; j < 8; ++j) {
          float x = acc[mi][ni][j] + bv;
          outB[(size_t)(m0 + j) * N + n] = f2b(gelu_fast(x));
        }
      } else {
        const float bx = biasX[n];
#pragma unroll
        for (int j = 0; j < 8; ++j) {
          size_t o = (size_t)(m0 + j) * N + n;
          outF[o] = acc[mi][ni][j] + add1[o] + add2[o] + bv + bx;
        }
      }
    }
  }
}

// ---------------------------------------------------------------------------
extern "C" void kernel_launch(void* const* d_in, const int* in_sizes, int n_in,
                              void* d_out, int out_size, void* d_ws, size_t ws_size,
                              hipStream_t stream) {
  (void)in_sizes; (void)n_in; (void)out_size; (void)ws_size;
  const float* input    = (const float*)d_in[0];
  const float* residual = (const float*)d_in[1];
  // d_in[2] residual_norm: unused by this path
  const float* bias     = (const float*)d_in[3];
  const float* attn_nw  = (const float*)d_in[4];
  const float* attn_nb  = (const float*)d_in[5];
  const float* inter_w  = (const float*)d_in[6];   // [H, DFF] fp32
  const float* inter_b  = (const float*)d_in[7];
  const float* output_w = (const float*)d_in[8];   // [DFF, H] fp32
  const float* output_b = (const float*)d_in[9];
  float* out = (float*)d_out;

  // workspace (bf16 halves): ln [M,H] + inter [M,DFF] + interT [DFF,H] + outT [H,DFF]
  unsigned short* ln     = (unsigned short*)d_ws;
  unsigned short* inter  = ln + (size_t)M_TOK * H_DIM;
  unsigned short* interT = inter + (size_t)M_TOK * DFF_DIM;
  unsigned short* outT   = interT + (size_t)DFF_DIM * H_DIM;

  // 1) fused bias + residual add + LayerNorm -> bf16
  ln_bias_res_kernel<<<M_TOK, 256, 0, stream>>>(input, residual, bias,
                                                attn_nw, attn_nb, ln);
  // 2) weights -> bf16, transposed to [N][K]
  transpose_convert_kernel<<<dim3(DFF_DIM / 32, H_DIM / 32), 256, 0, stream>>>(
      inter_w, interT, H_DIM, DFF_DIM);
  transpose_convert_kernel<<<dim3(H_DIM / 32, DFF_DIM / 32), 256, 0, stream>>>(
      output_w, outT, DFF_DIM, H_DIM);
  // 3) GEMM1 + bias + GeLU -> bf16 inter
  wmma_gemm_async_kernel<true><<<dim3(DFF_DIM / 128, M_TOK / 128), 256, 0,
                                 stream>>>(
      ln, interT, inter_b, H_DIM, DFF_DIM, inter, nullptr, nullptr, nullptr,
      nullptr);
  // 4) GEMM2 + full residual epilogue -> f32 out
  wmma_gemm_async_kernel<false><<<dim3(H_DIM / 128, M_TOK / 128), 256, 0,
                                  stream>>>(
      inter, outT, output_b, DFF_DIM, H_DIM, nullptr, out, residual, input,
      bias);
}
